// DynamicRoutingAggregator_12232066859206
// MI455X (gfx1250) — matrix operand order repaired and checked by reading, not compiled, not run on hardware
//
#include <hip/hip_runtime.h>
#include <hip/hip_bf16.h>
#include <stdint.h>

typedef __attribute__((ext_vector_type(16))) _Float16 v16h;
typedef __attribute__((ext_vector_type(8)))  _Float16 v8h;
typedef __attribute__((ext_vector_type(8)))  float    v8f;

#define CAPS   16
#define CDIM   64
#define BATCH  64
#define TOK    1024
#define DIN    512
#define NOUT   1024   // CAPS*CDIM

// ---- CDNA5 transcendental helpers (single hardware instructions) ----------
static __device__ __forceinline__ float fast_tanh(float x) {
#if __has_builtin(__builtin_amdgcn_tanhf)
    return __builtin_amdgcn_tanhf(x);        // v_tanh_f32
#else
    return tanhf(x);
#endif
}
static __device__ __forceinline__ float fast_exp(float x) {
    return __builtin_amdgcn_exp2f(x * 1.4426950408889634f);   // v_exp_f32
}
static __device__ __forceinline__ float fast_rcp(float x) {
    return __builtin_amdgcn_rcpf(x);         // v_rcp_f32
}
static __device__ __forceinline__ float fast_rsq(float x) {
    return __builtin_amdgcn_rsqf(x);         // v_rsq_f32
}

// ---------------------------------------------------------------------------
// Kernel 1: sharedH = tanh(X @ W + b) stored as f16.
// M=65536, K=512, N=1024.  Block tile 128x64, 8 waves, each wave 32x32
// as 2x2 v_wmma_f32_16x16x32_f16 tiles.  A/B converted f32->f16 into LDS.
// ---------------------------------------------------------------------------
__global__ __launch_bounds__(256)
void gemm_tanh_f16(const float* __restrict__ X, const float* __restrict__ W,
                   const float* __restrict__ bias, _Float16* __restrict__ out)
{
    __shared__ __align__(16) _Float16 ldsA[128 * 32];   // [m][k]
    __shared__ __align__(16) _Float16 ldsB[64 * 32];    // transposed: [n][k]

    const int tid  = threadIdx.x;
    const int lane = tid & 31;
    const int wid  = tid >> 5;
    const int wm   = (wid & 3) * 32;   // wave M offset within block tile
    const int wn   = (wid >> 2) * 32;  // wave N offset within block tile
    const int bm   = blockIdx.x * 128;
    const int bn   = blockIdx.y * 64;

    v8f acc[2][2] = {};                // zero accumulators

    const int ar = tid >> 1;           // 0..127 : A row within tile
    const int ac = (tid & 1) * 16;     // 0 / 16 : A col half
    const int kp = tid & 15;           // k-pair index 0..15 (k = 2*kp)
    const int nc = (tid >> 4) * 4;     // 0..60  : W col group of 4

    const int rsel = lane >> 4;        // 0/1 : lane half
    const int rlo  = lane & 15;

    for (int ks = 0; ks < DIN; ks += 32) {
        // ---- stage A tile 128x32 (f32 -> f16), two ds_store_b128 ----
        {
            const float4* ap = (const float4*)(X + (size_t)(bm + ar) * DIN + ks + ac);
            float4 x0 = ap[0], x1 = ap[1], x2 = ap[2], x3 = ap[3];
            v8h h0, h1;
            h0[0]=(_Float16)x0.x; h0[1]=(_Float16)x0.y; h0[2]=(_Float16)x0.z; h0[3]=(_Float16)x0.w;
            h0[4]=(_Float16)x1.x; h0[5]=(_Float16)x1.y; h0[6]=(_Float16)x1.z; h0[7]=(_Float16)x1.w;
            h1[0]=(_Float16)x2.x; h1[1]=(_Float16)x2.y; h1[2]=(_Float16)x2.z; h1[3]=(_Float16)x2.w;
            h1[4]=(_Float16)x3.x; h1[5]=(_Float16)x3.y; h1[6]=(_Float16)x3.z; h1[7]=(_Float16)x3.w;
            *(v8h*)&ldsA[ar * 32 + ac]     = h0;
            *(v8h*)&ldsA[ar * 32 + ac + 8] = h1;
        }
        // ---- stage B tile 32x64, transposed into [n][k], packed f16 pairs ----
        {
            const float4* wp0 = (const float4*)(W + (size_t)(ks + 2 * kp)     * NOUT + bn + nc);
            const float4* wp1 = (const float4*)(W + (size_t)(ks + 2 * kp + 1) * NOUT + bn + nc);
            float4 w0 = wp0[0], w1 = wp1[0];
            uint32_t* lb = (uint32_t*)ldsB;                 // [n][k/2] u32 view
            union { _Float16 h[2]; uint32_t u; } p;
            p.h[0] = (_Float16)w0.x; p.h[1] = (_Float16)w1.x; lb[(nc + 0) * 16 + kp] = p.u;
            p.h[0] = (_Float16)w0.y; p.h[1] = (_Float16)w1.y; lb[(nc + 1) * 16 + kp] = p.u;
            p.h[0] = (_Float16)w0.z; p.h[1] = (_Float16)w1.z; lb[(nc + 2) * 16 + kp] = p.u;
            p.h[0] = (_Float16)w0.w; p.h[1] = (_Float16)w1.w; lb[(nc + 3) * 16 + kp] = p.u;
        }
        __syncthreads();

        // ---- load fragments (ISA 7.12.2 16-bit layouts) and issue WMMA ----
        v16h afrag[2], bfrag[2];
#pragma unroll
        for (int tm = 0; tm < 2; ++tm) {
            // A: lane rlo = row; elems 0..7 -> K = rsel*8 + 0..7 ; elems 8..15 -> +16
            const uint32_t* p0 = (const uint32_t*)&ldsA[(wm + tm * 16 + rlo) * 32 + rsel * 8];
            const uint32_t* p1 = (const uint32_t*)&ldsA[(wm + tm * 16 + rlo) * 32 + rsel * 8 + 16];
            union { v16h v; uint32_t u[8]; } fa;
#pragma unroll
            for (int j = 0; j < 4; ++j) { fa.u[j] = p0[j]; fa.u[4 + j] = p1[j]; }
            afrag[tm] = fa.v;
        }
#pragma unroll
        for (int tn = 0; tn < 2; ++tn) {
            // B: lane rlo = col; elems 0..7 -> K = rsel*8 + 0..7 ; elems 8..15 -> +16
            const uint32_t* p0 = (const uint32_t*)&ldsB[(wn + tn * 16 + rlo) * 32 + rsel * 8];
            const uint32_t* p1 = (const uint32_t*)&ldsB[(wn + tn * 16 + rlo) * 32 + rsel * 8 + 16];
            union { v16h v; uint32_t u[8]; } fb;
#pragma unroll
            for (int j = 0; j < 4; ++j) { fb.u[j] = p0[j]; fb.u[4 + j] = p1[j]; }
            bfrag[tn] = fb.v;
        }
#pragma unroll
        for (int tm = 0; tm < 2; ++tm)
#pragma unroll
            for (int tn = 0; tn < 2; ++tn)
                acc[tm][tn] = __builtin_amdgcn_wmma_f32_16x16x32_f16(
                    false, afrag[tm], false, bfrag[tn], (short)0, acc[tm][tn], false, false);
        __syncthreads();
    }

    // ---- epilogue: bias + hardware tanh, store f16.  C/D: vgpr j -> row rsel*8+j
#pragma unroll
    for (int tm = 0; tm < 2; ++tm)
#pragma unroll
        for (int tn = 0; tn < 2; ++tn) {
            const int gn = bn + wn + tn * 16 + rlo;
            const float bv = bias[gn];
#pragma unroll
            for (int j = 0; j < 8; ++j) {
                const int gm = bm + wm + tm * 16 + rsel * 8 + j;
                out[(size_t)gm * NOUT + gn] = (_Float16)fast_tanh(acc[tm][tn][j] + bv);
            }
        }
}

// ---------------------------------------------------------------------------
// Kernel 2: iteration-1 S (softmax of zero logits is uniform 1/16, x mask)
// grid (CAPS, BATCH), block 64 (= d)
// ---------------------------------------------------------------------------
__global__ void s1_kernel(const _Float16* __restrict__ sh, const int* __restrict__ mask,
                          float* __restrict__ S)
{
    const int c = blockIdx.x, b = blockIdx.y, d = threadIdx.x;
    const int* mp = mask + b * TOK;
    float acc = 0.f;
    for (int t = 0; t < TOK; ++t) {
        const float v = (float)sh[(((size_t)b * TOK + t) * CAPS + c) * CDIM + d];
        acc = fmaf((float)mp[t], v, acc);       // branchless mask
    }
    S[(b * CAPS + c) * CDIM + d] = acc * (1.0f / 16.0f);
}

// ---------------------------------------------------------------------------
// Kernel 3: V = squash(S) over the 64-dim axis.  grid 1024 (= b*caps), block 64
// ---------------------------------------------------------------------------
__global__ void squash_kernel(const float* __restrict__ S, float* __restrict__ V)
{
    const int g = blockIdx.x, d = threadIdx.x;
    __shared__ float red[64];
    const float s = S[g * CDIM + d];
    red[d] = s * s;
    __syncthreads();
    for (int off = 32; off > 0; off >>= 1) {
        if (d < off) red[d] += red[d + off];
        __syncthreads();
    }
    const float ns = red[0];
    V[g * CDIM + d] = ns * fast_rcp(1.f + ns) * s * fast_rsq(ns + 1e-8f);
}

// ---------------------------------------------------------------------------
// Kernel 4: fused routing pass:  lnew = logits + shared.V ; C = softmax(lnew)*mask ;
// S += C * shared ; logits <- lnew.   Wave-per-token, lane = (cap, half-of-64).
// grid (8, BATCH), block 256 (8 waves x 16 tokens each).
// ---------------------------------------------------------------------------
__global__ __launch_bounds__(256)
void route_kernel(const _Float16* __restrict__ sh, const int* __restrict__ mask,
                  const float* __restrict__ V, float* __restrict__ logits,
                  float* __restrict__ S)
{
    __shared__ float Vl[NOUT];
    const int b = blockIdx.y;
    const int tid = threadIdx.x;
    for (int i = tid; i < NOUT; i += 256) Vl[i] = V[b * NOUT + i];
    __syncthreads();

    const int lane = tid & 31;
    const int wid  = tid >> 5;
    const int c = lane >> 1;       // capsule 0..15
    const int h = lane & 1;        // which 32-dim half
    const float* vrow = &Vl[c * CDIM + h * 32];

    float acc[32];
#pragma unroll
    for (int j = 0; j < 32; ++j) acc[j] = 0.f;

    const int t0 = blockIdx.x * 128;
    for (int t = t0 + wid; t < t0 + 128; t += 8) {
        const size_t base = (((size_t)b * TOK + t) * CAPS + c) * CDIM + h * 32;
        const v8h* sp = (const v8h*)(sh + base);
        v8h q0 = sp[0], q1 = sp[1], q2 = sp[2], q3 = sp[3];
        float vals[32];
#pragma unroll
        for (int j = 0; j < 8; ++j) {
            vals[j]      = (float)q0[j];
            vals[8 + j]  = (float)q1[j];
            vals[16 + j] = (float)q2[j];
            vals[24 + j] = (float)q3[j];
        }
        float r = 0.f;
#pragma unroll
        for (int j = 0; j < 32; ++j) r += vals[j] * vrow[j];
        r += __shfl_xor(r, 1);                         // combine halves -> full 64-dot

        const int lidx = (b * TOK + t) * CAPS + c;
        const float lnew = logits[lidx] + r;
        if (h == 0) logits[lidx] = lnew;

        const float mk = (float)mask[b * TOK + t];

        // softmax over 16 caps (value duplicated across lane pairs)
        float mx = lnew;
#pragma unroll
        for (int off = 2; off < 32; off <<= 1) mx = fmaxf(mx, __shfl_xor(mx, off));
        const float e = fast_exp(lnew - mx);
        float ssum = e;
#pragma unroll
        for (int off = 1; off < 32; off <<= 1) ssum += __shfl_xor(ssum, off);  // = 2*sum
        const float Cc = 2.0f * e * fast_rcp(ssum) * mk;

#pragma unroll
        for (int j = 0; j < 32; ++j) acc[j] += Cc * vals[j];
    }

    float* spS = &S[(b * CAPS + c) * CDIM + h * 32];
#pragma unroll
    for (int j = 0; j < 32; ++j) atomicAdd(&spS[j], acc[j]);
}

__global__ void zero_kernel(float* __restrict__ p, int n)
{
    const int i = blockIdx.x * 256 + threadIdx.x;
    if (i < n) p[i] = 0.f;
}

// ---------------------------------------------------------------------------
extern "C" void kernel_launch(void* const* d_in, const int* in_sizes, int n_in,
                              void* d_out, int out_size, void* d_ws, size_t ws_size,
                              hipStream_t stream)
{
    const float* X    = (const float*)d_in[0];   // [64,1024,512]
    const int*   mask = (const int*)  d_in[1];   // [64,1024]
    const float* W    = (const float*)d_in[2];   // [512,1024]
    const float* bias = (const float*)d_in[3];   // [1024]
    float*       out  = (float*)d_out;           // [64,1024]

    char* ws = (char*)d_ws;
    const size_t nShared = (size_t)BATCH * TOK * NOUT;      // 67,108,864 f16 = 128 MB
    const size_t nLogits = (size_t)BATCH * TOK * CAPS;      // 1,048,576 f32 = 4 MB
    const size_t nS      = (size_t)BATCH * NOUT;            // 65,536 f32

    _Float16* sharedH = (_Float16*)ws;
    float*    logits  = (float*)(ws + nShared * sizeof(_Float16));
    float*    S       = (float*)(ws + nShared * sizeof(_Float16) + nLogits * sizeof(float));
    float*    V       = S + nS;

    // init logits (the -1e32 pin is equivalent to the *mask factor, so start at 0)
    zero_kernel<<<(int)((nLogits + 255) / 256), 256, 0, stream>>>(logits, (int)nLogits);

    // shared = tanh(X @ W + b), stored f16
    gemm_tanh_f16<<<dim3(512, 16), 256, 0, stream>>>(X, W, bias, sharedH);

    // iteration 1 (uniform C)
    s1_kernel<<<dim3(CAPS, BATCH), CDIM, 0, stream>>>(sharedH, mask, S);
    squash_kernel<<<BATCH * CAPS, CDIM, 0, stream>>>(S, V);

    // iteration 2 (fused logit update + softmax + accumulate)
    zero_kernel<<<(int)((nS + 255) / 256), 256, 0, stream>>>(S, (int)nS);
    route_kernel<<<dim3(8, BATCH), 256, 0, stream>>>(sharedH, mask, V, logits, S);
    squash_kernel<<<BATCH * CAPS, CDIM, 0, stream>>>(S, V);

    // iteration 3
    zero_kernel<<<(int)((nS + 255) / 256), 256, 0, stream>>>(S, (int)nS);
    route_kernel<<<dim3(8, BATCH), 256, 0, stream>>>(sharedH, mask, V, logits, S);
    squash_kernel<<<BATCH * CAPS, CDIM, 0, stream>>>(S, out);
}
